// SelfAttention_31739808317917
// MI455X (gfx1250) — matrix operand order repaired and checked by reading, not compiled, not run on hardware
//
#include <hip/hip_runtime.h>
#include <hip/hip_bf16.h>

// ---------------------------------------------------------------------------
// Self-attention (B=4, C=256, H=W=64 -> N=4096, IC=32) for gfx1250 (CDNA5).
// Kernel 0: convert Wq/Wk/Wv f32 -> one f16 weight block Wall[320][256].
// Kernel 1: fused q/k/v 1x1-conv projections via v_wmma_f32_16x16x32_f16.
// Kernel 2: flash-attention (online softmax), WMMA throughout; K/V tiles
//           double-buffered in LDS via global_load_async_to_lds_b64/b128
//           (ASYNCcnt), staging latency hidden behind WMMA compute.
// Fragments for each K-chain are preloaded into distinct registers so loads
// batch into one clause + one wait, and WMMAs chain only on the accumulator.
// ---------------------------------------------------------------------------

typedef __attribute__((ext_vector_type(16))) _Float16 v16h;
typedef __attribute__((ext_vector_type(8)))  float    v8f;

union F16x16 {
    v16h v;
    _Float16 h[16];
    unsigned int u[8];
};

static constexpr int BB  = 4;
static constexpr int CC  = 256;
static constexpr int NN  = 4096;   // H*W
static constexpr int ICC = 32;

// A-matrix (f16 16x32, M=lane%16) per-lane K offset of half-pair v (even).
__device__ __forceinline__ int akoff(int lane, int v) {
    int g = (lane >> 4) << 3;
    return (v < 4) ? (g + 2 * v) : (16 + g + 2 * (v - 4));
}
// B-matrix (f16 32x16, N=lane%16): same K order as akoff.
__device__ __forceinline__ int bkoff(int lane, int v) {
    int g = (lane >> 4) << 3;
    return (v < 4) ? (g + 2 * v) : (16 + g + 2 * (v - 4));
}

// Async global->LDS copies (CDNA5, ASYNCcnt).
__device__ __forceinline__ void async_g2l_b128(unsigned lds, unsigned long long ga) {
    asm volatile("global_load_async_to_lds_b128 %0, %1, off"
                 :: "v"(lds), "v"(ga) : "memory");
}
__device__ __forceinline__ void async_g2l_b64(unsigned lds, unsigned long long ga) {
    asm volatile("global_load_async_to_lds_b64 %0, %1, off"
                 :: "v"(lds), "v"(ga) : "memory");
}
__device__ __forceinline__ void wait_async0() {
    asm volatile("s_wait_asynccnt 0x0" ::: "memory");
}
__device__ __forceinline__ unsigned lds_addr(const void* p) {
    return (unsigned)(uintptr_t)p;     // LDS aperture: low 32 bits = LDS offset
}

// ---------------------------------------------------------------------------
// Kernel 0: weights f32 -> f16, packed as Wall[320][256] (q:0-31,k:32-63,v:64-319)
// ---------------------------------------------------------------------------
__global__ __launch_bounds__(256) void wcvt_kernel(
    const float* __restrict__ Wq, const float* __restrict__ Wk,
    const float* __restrict__ Wv, _Float16* __restrict__ Wall)
{
    const int idx = blockIdx.x * 256 + threadIdx.x;   // 0 .. 320*256-1
    const int row = idx >> 8;
    const int col = idx & 255;
    float v;
    if (row < 32)       v = Wq[row * 256 + col];
    else if (row < 64)  v = Wk[(row - 32) * 256 + col];
    else                v = Wv[(row - 64) * 256 + col];
    Wall[idx] = (_Float16)v;
}

// ---------------------------------------------------------------------------
// Kernel 1: projections.  grid = B * (N/64), block = 256 (8 waves).
// ---------------------------------------------------------------------------
static constexpr int XTS = 264;    // LDS stride (halves) for x tile, padded

__global__ __launch_bounds__(256) void proj_kernel(
    const float* __restrict__ x,
    const _Float16* __restrict__ Wall,
    const float* __restrict__ bq, const float* __restrict__ bk,
    const float* __restrict__ bv,
    _Float16* __restrict__ qT, _Float16* __restrict__ kT,
    _Float16* __restrict__ vW)
{
    __shared__ __align__(16) _Float16 xt[64 * XTS];   // [n:64][c:256] f16

    const int tid  = threadIdx.x;
    const int lane = tid & 31;
    const int wave = __builtin_amdgcn_readfirstlane(tid >> 5);   // SGPR-uniform
    const int bb   = blockIdx.x >> 6;          // batch
    const int nb   = (blockIdx.x & 63) << 6;   // pixel-tile base

    // Stage + transpose + f32->f16 convert the x tile: [C=256][n=64] -> [n][c]
    const float* xb = x + (size_t)bb * CC * NN;
    for (int idx = tid; idx < 64 * 256; idx += 256) {
        int c = idx >> 6;
        int n = idx & 63;
        xt[n * XTS + c] = (_Float16)xb[(size_t)c * NN + nb + n];
    }
    __syncthreads();

    const int m   = lane & 15;
    const int rlo = (lane >> 4) << 3;          // 0 or 8 (row-half base)

    // 20 row-tiles (q:0-1, k:2-3, v:4-19) x 4 n-tiles = 80 WMMA output tiles.
    for (int tile = wave; tile < 80; tile += 8) {
        const int rt  = tile >> 2;             // scalar
        const int nt  = tile & 3;              // scalar
        const int rog = rt * 16;               // scalar row base in Wall

        // Preload ALL fragments for the K=256 chain, then 8 chained WMMAs.
        F16x16 a[8], b[8];
        const _Float16* wrow = Wall + (size_t)(rog + m) * CC;
        const _Float16* xrow = &xt[(nt * 16 + m) * XTS];
#pragma unroll
        for (int kb = 0; kb < 8; ++kb) {
#pragma unroll
            for (int v = 0; v < 8; ++v)
                a[kb].u[v] = *(const unsigned int*)(wrow + kb * 32 + akoff(lane, v));
#pragma unroll
            for (int v = 0; v < 8; ++v)
                b[kb].u[v] = *(const unsigned int*)(xrow + kb * 32 + bkoff(lane, v));
        }
        v8f acc = {};
#pragma unroll
        for (int kb = 0; kb < 8; ++kb)
            acc = __builtin_amdgcn_wmma_f32_16x16x32_f16(
                false, a[kb].v, false, b[kb].v, (short)0, acc, false, false);

        // One scalar branch per tile: [rog, rog+16) never straddles 32/64.
        const int ncol = nb + nt * 16 + m;
        if (rog < 32) {
            _Float16* dst = qT + ((size_t)bb * NN + ncol) * ICC + rog;
#pragma unroll
            for (int j = 0; j < 8; ++j)
                dst[j + rlo] = (_Float16)(acc[j] + bq[rog + j + rlo]);
        } else if (rog < 64) {
            _Float16* dst = kT + ((size_t)bb * NN + ncol) * ICC + (rog - 32);
#pragma unroll
            for (int j = 0; j < 8; ++j)
                dst[j + rlo] = (_Float16)(acc[j] + bk[rog - 32 + j + rlo]);
        } else {
#pragma unroll
            for (int j = 0; j < 8; ++j) {
                const int c = rog - 64 + j + rlo;
                vW[((size_t)bb * CC + c) * NN + ncol] =
                    (_Float16)(acc[j] + bv[c]);
            }
        }
    }
}

// ---------------------------------------------------------------------------
// Kernel 2: flash attention.  grid = B * (N/64), block = 256 (8 waves).
// wave -> (query tile qt = wave/2 of 16 rows, channel half ch0 = (wave&1)*128).
// K/V tiles double-buffered; async staging of tile i+1 overlaps compute on i.
// ---------------------------------------------------------------------------
static constexpr int KTS = 40;     // K tile LDS stride (halves); 80 B rows
static constexpr int VTS = 72;     // V / P tile LDS stride (halves); 144 B rows
static constexpr int KTILE_H = 64 * KTS;      // halves per K buffer
static constexpr int VTILE_H = 256 * VTS;     // halves per V buffer

__device__ __forceinline__ void stage_tiles(const char* gk, const char* gv,
                                            _Float16* KtB, _Float16* VtB,
                                            int tid)
{
#pragma unroll
    for (int it = 0; it < 2; ++it) {                   // K: 64 rows x 64 B
        const int idx = tid + it * 256;                // 0..511
        const int row = idx >> 3, seg = idx & 7;
        async_g2l_b64(lds_addr((const char*)KtB + row * (KTS * 2) + seg * 8),
                      (unsigned long long)(uintptr_t)(gk + row * 64 + seg * 8));
    }
#pragma unroll
    for (int it = 0; it < 8; ++it) {                   // V: 256 rows x 128 B
        const int idx = tid + it * 256;                // 0..2047
        const int row = idx >> 3, seg = idx & 7;
        async_g2l_b128(lds_addr((const char*)VtB + row * (VTS * 2) + seg * 16),
                       (unsigned long long)(uintptr_t)(gv + (size_t)row * (NN * 2) + seg * 16));
    }
}

__global__ __launch_bounds__(256) void attn_kernel(
    const float* __restrict__ x,
    const _Float16* __restrict__ qT,
    const _Float16* __restrict__ kT,
    const _Float16* __restrict__ vW,
    const float* __restrict__ gamma_p,
    float* __restrict__ out)
{
    __shared__ __align__(16) _Float16 Kt[2 * KTILE_H];    // 2 x [key:64][ic:32]
    __shared__ __align__(16) _Float16 Vt[2 * VTILE_H];    // 2 x [ch:256][key:64]
    __shared__ __align__(16) _Float16 Pt[4 * 16 * VTS];   // [qt][row:16][key:64]

    const int tid  = threadIdx.x;
    const int lane = tid & 31;
    const int wave = __builtin_amdgcn_readfirstlane(tid >> 5);   // SGPR-uniform
    const int bb   = blockIdx.x >> 6;
    const int qb   = (blockIdx.x & 63) << 6;
    const int qt   = wave >> 1;                 // scalar
    const int ch0  = (wave & 1) * 128;          // scalar
    const int m    = lane & 15;
    const float gamma = gamma_p[0];

    // Q A-fragment for this wave's 16 query rows (resident for whole kernel).
    F16x16 qa;
    {
        const _Float16* qrow = qT + ((size_t)bb * NN + qb + qt * 16 + m) * ICC;
#pragma unroll
        for (int v = 0; v < 8; ++v)
            qa.u[v] = *(const unsigned int*)(qrow + akoff(lane, v));
    }

    v8f o[8];
#pragma unroll
    for (int c = 0; c < 8; ++c) o[c] = (v8f){};
    float mrow[8], lrow[8];
#pragma unroll
    for (int j = 0; j < 8; ++j) { mrow[j] = -1e30f; lrow[j] = 0.0f; }

    const char* gkbase = (const char*)(kT + ((size_t)bb * NN) * ICC);
    const char* gvbase = (const char*)(vW + (size_t)bb * CC * NN);

    // Prologue: stage tile 0 into buffer 0.
    stage_tiles(gkbase, gvbase, Kt, Vt, tid);
    wait_async0();
    __syncthreads();

    constexpr int NIT = NN / 64;   // 64 key tiles
    for (int kt = 0; kt < NIT; ++kt) {
        _Float16* Kc = Kt + (kt & 1) * KTILE_H;
        _Float16* Vc = Vt + (kt & 1) * VTILE_H;

        // Kick off async staging of the NEXT tile into the other buffer.
        if (kt + 1 < NIT) {
            stage_tiles(gkbase + (size_t)(kt + 1) * 64 * ICC * 2,
                        gvbase + (size_t)(kt + 1) * 64 * 2,
                        Kt + ((kt + 1) & 1) * KTILE_H,
                        Vt + ((kt + 1) & 1) * VTILE_H, tid);
            if (kt + 2 < NIT) {    // nudge tile kt+2 toward L2 (global_prefetch_b8)
                __builtin_prefetch(gkbase + (size_t)(kt + 2) * 64 * ICC * 2, 0, 0);
                __builtin_prefetch(gvbase + (size_t)(kt + 2) * 64 * 2, 0, 0);
            }
        }

        // ---- scores S = Q(16x32) * K^T(32x64): preload 4 B-frags, 4 WMMAs ----
        F16x16 kf[4];
#pragma unroll
        for (int t = 0; t < 4; ++t) {
            const _Float16* krow = &Kc[(t * 16 + m) * KTS];
#pragma unroll
            for (int v = 0; v < 8; ++v)
                kf[t].u[v] = *(const unsigned int*)(krow + bkoff(lane, v));
        }
        v8f s[4];
#pragma unroll
        for (int t = 0; t < 4; ++t) {
            v8f z = {};
            s[t] = __builtin_amdgcn_wmma_f32_16x16x32_f16(
                false, qa.v, false, kf[t].v, (short)0, z, false, false);
        }

        // ---- online softmax (row r lives in lanes of one 16-lane half) ----
#pragma unroll
        for (int j = 0; j < 8; ++j) {
            float mv = fmaxf(fmaxf(s[0][j], s[1][j]), fmaxf(s[2][j], s[3][j]));
#pragma unroll
            for (int d = 1; d < 16; d <<= 1)
                mv = fmaxf(mv, __shfl_xor(mv, d, 16));
            const float mnew  = fmaxf(mrow[j], mv);
            const float alpha = __expf(mrow[j] - mnew);
            mrow[j] = mnew;
            float ps = 0.0f;
#pragma unroll
            for (int t = 0; t < 4; ++t) {
                const float p = __expf(s[t][j] - mnew);
                s[t][j] = p;
                ps += p;
            }
            lrow[j] = lrow[j] * alpha + ps;
#pragma unroll
            for (int c = 0; c < 8; ++c) o[c][j] *= alpha;
        }

        // ---- P (C-layout) -> LDS (A-layout source); one wave per qtile ----
        if (ch0 == 0) {                         // scalar branch
            _Float16* prow = &Pt[qt * 16 * VTS];
#pragma unroll
            for (int t = 0; t < 4; ++t)
#pragma unroll
                for (int j = 0; j < 8; ++j)
                    prow[(j + ((lane >> 4) << 3)) * VTS + t * 16 + m] =
                        (_Float16)s[t][j];
        }
        __syncthreads();

        // ---- O(16x128) += P(16x64) * V^T(64x128): preload frags, 8 WMMAs/kk ----
#pragma unroll
        for (int kk = 0; kk < 2; ++kk) {
            F16x16 pa;
            const _Float16* pr = &Pt[(qt * 16 + m) * VTS + kk * 32];
#pragma unroll
            for (int v = 0; v < 8; ++v)
                pa.u[v] = *(const unsigned int*)(pr + akoff(lane, v));
            F16x16 vb[8];
#pragma unroll
            for (int c = 0; c < 8; ++c) {
                const _Float16* vr = &Vc[(ch0 + c * 16 + m) * VTS + kk * 32];
#pragma unroll
                for (int v = 0; v < 8; ++v)
                    vb[c].u[v] = *(const unsigned int*)(vr + bkoff(lane, v));
            }
#pragma unroll
            for (int c = 0; c < 8; ++c)
                o[c] = __builtin_amdgcn_wmma_f32_16x16x32_f16(
                    false, pa.v, false, vb[c].v, (short)0, o[c], false, false);
        }

        // Next-tile asyncs must have landed before anyone reads that buffer.
        wait_async0();
        __syncthreads();
    }

    // ---- epilogue: 1/l, gamma, residual ----
#pragma unroll
    for (int j = 0; j < 8; ++j) {
        float lv = lrow[j];
#pragma unroll
        for (int d = 1; d < 16; d <<= 1) lv += __shfl_xor(lv, d, 16);
        lrow[j] = 1.0f / lv;
    }
#pragma unroll
    for (int c = 0; c < 8; ++c) {
        const int ch = ch0 + c * 16 + m;
#pragma unroll
        for (int j = 0; j < 8; ++j) {
            const int nq = qb + qt * 16 + j + ((lane >> 4) << 3);
            const size_t idx = ((size_t)bb * CC + ch) * NN + nq;
            out[idx] = gamma * (o[c][j] * lrow[j]) + x[idx];
        }
    }
}

// ---------------------------------------------------------------------------
extern "C" void kernel_launch(void* const* d_in, const int* in_sizes, int n_in,
                              void* d_out, int out_size, void* d_ws, size_t ws_size,
                              hipStream_t stream) {
    (void)in_sizes; (void)n_in; (void)out_size; (void)ws_size;
    const float* x     = (const float*)d_in[0];
    const float* Wq    = (const float*)d_in[1];
    const float* bq    = (const float*)d_in[2];
    const float* Wk    = (const float*)d_in[3];
    const float* bk    = (const float*)d_in[4];
    const float* Wv    = (const float*)d_in[5];
    const float* bv    = (const float*)d_in[6];
    const float* gamma = (const float*)d_in[7];
    float* out = (float*)d_out;

    _Float16* qT   = (_Float16*)d_ws;                     // B*N*32  f16 (1 MB)
    _Float16* kT   = qT + (size_t)BB * NN * ICC;          // B*N*32  f16 (1 MB)
    _Float16* vW   = kT + (size_t)BB * NN * ICC;          // B*C*N   f16 (8 MB)
    _Float16* Wall = vW + (size_t)BB * CC * NN;           // 320*256 f16 (160 KB)

    wcvt_kernel<<<(320 * 256) / 256, 256, 0, stream>>>(Wq, Wk, Wv, Wall);

    const int blocks = BB * (NN / 64);                    // 256
    proj_kernel<<<blocks, 256, 0, stream>>>(x, Wall, bq, bk, bv, qT, kT, vW);
    attn_kernel<<<blocks, 256, 0, stream>>>(x, qT, kT, vW, gamma, out);
}
